// HANModel_50079318671417
// MI455X (gfx1250) — matrix operand order repaired and checked by reading, not compiled, not run on hardware
//
#include <hip/hip_runtime.h>
#include <math.h>
#include <limits.h>

// ---------------------------------------------------------------------------
// HAN forward for MI455X (gfx1250, wave32, WMMA).
//   1) pack W (fp32 -> bf16, native cvt) into WMMA B-fragments    [prepack]
//   2) h = x @ W + b via v_wmma_f32_16x16x32_bf16                 [gemm_bias]
//      32x128 C tile / block, K=64 slabs, double-buffered LDS,
//      all fragments loaded before 4 back-to-back WMMAs per sync
//   3) fused per-head attention scalars (one pass over h)         [headdot*]
//   4) edge softmax: atomic segment-max (encoded int), denom,     [att_*]
//      alpha computed once per head half-wave + __shfl broadcast
//   5) relu, semantic score GEMM (WMMA + tanh/q reduction)        [gemm_score]
//   6) softmax(beta), fuse, ELU, 128->4 head                      [final]
// ---------------------------------------------------------------------------

#define N_NEWS  30000
#define N_INTER 60000
#define F_IN    768
#define HID     128
#define NHEAD   8
#define HDIM    16
#define C_OUT   4

#define LDS_PITCH 72   // elements per row (64 data + 8 pad); 144B: 16B-aligned,
                       // row*36 DWORD-banks mod 64 distinct for 16-row groups

typedef __attribute__((ext_vector_type(16))) __bf16 v16bf;
typedef __attribute__((ext_vector_type(2)))  __bf16 v2bf;
typedef __attribute__((ext_vector_type(2)))  float  v2f;
typedef __attribute__((ext_vector_type(8)))  float  v8f;

union Frag16 { uint4 u[2]; v16bf v; };

// pack two floats to packed bf16: single v_cvt_pk_bf16_f32
__device__ __forceinline__ unsigned pack2bf(float lo, float hi) {
    v2f f;
    f.x = lo; f.y = hi;
    union { v2bf v; unsigned u; } cv;
    cv.v = __builtin_convertvector(f, v2bf);
    return cv.u;
}

// order-preserving float<->int encoding for atomicMax on float values
__device__ __forceinline__ int encF(float v) {
    int b = __float_as_int(v);
    return b >= 0 ? b : (b ^ 0x7FFFFFFF);
}
__device__ __forceinline__ float decF(int v) {
    int b = v >= 0 ? v : (v ^ 0x7FFFFFFF);
    return __int_as_float(b);
}

__device__ __forceinline__ v16bf load_afrag(const __bf16* p) {
    Frag16 a;
    a.u[0] = *reinterpret_cast<const uint4*>(p);
    a.u[1] = *reinterpret_cast<const uint4*>(p + 16);
    return a.v;
}
__device__ __forceinline__ v16bf load_bfrag(const __bf16* Bp, int kt, int w, int lane) {
    const uint4* bp = reinterpret_cast<const uint4*>(
        Bp + (((size_t)kt * 8 + w) * 32 + lane) * 16);
    Frag16 b;
    b.u[0] = bp[0];
    b.u[1] = bp[1];
    return b.v;
}

// ---------------------------------------------------------------------------
// Pre-pack W [K x 128] fp32 (row-major) into bf16 WMMA B-fragments.
// lane half h: elem j<8 -> K = ktile*32 + 8h + j, elem j>=8 -> K = ktile*32 + 16 + 8h + (j-8),
// column = lane & 15. dst index: (((ktile*8 + coltile)*32 + lane)*16 + j)
// ---------------------------------------------------------------------------
__global__ void prepack_w(const float* __restrict__ W, __bf16* __restrict__ dst, int K) {
    int kt = blockIdx.x >> 3;
    int ct = blockIdx.x & 7;
    int lane = threadIdx.x;
    int col = lane & 15;
    int kg  = (lane >> 4) * 8;
    size_t base = (((size_t)(kt * 8 + ct)) * 32 + lane) * 16;
    for (int j = 0; j < 16; ++j) {
        float v = W[(size_t)(kt * 32 + kg + j + (j >= 8 ? 8 : 0)) * HID + ct * 16 + col];
        dst[base + j] = (__bf16)v;
    }
}

// stage 8 floats -> 8 bf16 -> one b128 LDS store (4x v_cvt_pk_bf16_f32)
__device__ __forceinline__ void stage8(__bf16* dst, float4 f0, float4 f1) {
    uint4 u;
    u.x = pack2bf(f0.x, f0.y); u.y = pack2bf(f0.z, f0.w);
    u.z = pack2bf(f1.x, f1.y); u.w = pack2bf(f1.z, f1.w);
    *reinterpret_cast<uint4*>(dst) = u;
}

// ---------------------------------------------------------------------------
// C[M x 128] = A[M x K] @ Wpacked + bias.  block = 256 thr (8 waves),
// 32x128 tile / block; wave w -> cols [16w,16w+16), two 16-row accumulators.
// ---------------------------------------------------------------------------
__global__ void __launch_bounds__(256)
gemm_bias(const float* __restrict__ A, const __bf16* __restrict__ Bp,
          const float* __restrict__ bias, float* __restrict__ C, int K, int M) {
    __shared__ __bf16 lds[2][32 * LDS_PITCH];
    const int tid = threadIdx.x;
    const int w = tid >> 5, lane = tid & 31;
    const int rowbase = blockIdx.x * 32;
    const int col = lane & 15;
    const int off = (lane < 16) ? 0 : 8;        // A-fragment K-group select
    const int a0row = lane & 15, a1row = 16 + (lane & 15);
    const int lrow = tid >> 3;                   // coop-load row 0..31
    const int lk   = (tid & 7) * 8;              // coop-load k (8 floats/thread)
    const int nslab = K >> 6;                    // K=64 per stage

    int grow = rowbase + lrow;
    if (grow > M - 1) grow = M - 1;              // clamp (stores are masked)
    const float* aptr = A + (size_t)grow * K + lk;

    {   // prologue: stage slab 0
        float4 f0 = *reinterpret_cast<const float4*>(aptr);
        float4 f1 = *reinterpret_cast<const float4*>(aptr + 4);
        stage8(&lds[0][lrow * LDS_PITCH + lk], f0, f1);
    }
    __syncthreads();

    v8f c0 = {}, c1 = {};
    for (int s = 0; s < nslab; ++s) {
        const int cur = s & 1;
        const bool more = (s + 1) < nslab;
        float4 g0, g1;
        if (more) {                               // issue next-slab A loads early
            const float* p = aptr + (size_t)(s + 1) * 64;
            g0 = *reinterpret_cast<const float4*>(p);
            g1 = *reinterpret_cast<const float4*>(p + 4);
        }
        // load all fragments, then 4 back-to-back WMMAs
        const __bf16* base0 = &lds[cur][0] + off;
        v16bf a00 = load_afrag(base0 + a0row * LDS_PITCH);
        v16bf a01 = load_afrag(base0 + a1row * LDS_PITCH);
        v16bf a10 = load_afrag(base0 + 32 + a0row * LDS_PITCH);
        v16bf a11 = load_afrag(base0 + 32 + a1row * LDS_PITCH);
        v16bf b0  = load_bfrag(Bp, s * 2 + 0, w, lane);
        v16bf b1  = load_bfrag(Bp, s * 2 + 1, w, lane);
        c0 = __builtin_amdgcn_wmma_f32_16x16x32_bf16(false, a00, false, b0,
                                                     (short)0, c0, false, false);
        c1 = __builtin_amdgcn_wmma_f32_16x16x32_bf16(false, a01, false, b0,
                                                     (short)0, c1, false, false);
        c0 = __builtin_amdgcn_wmma_f32_16x16x32_bf16(false, a10, false, b1,
                                                     (short)0, c0, false, false);
        c1 = __builtin_amdgcn_wmma_f32_16x16x32_bf16(false, a11, false, b1,
                                                     (short)0, c1, false, false);
        if (more)                                 // stage next slab (other buffer)
            stage8(&lds[cur ^ 1][lrow * LDS_PITCH + lk], g0, g1);
        __syncthreads();
    }

    const int cg = w * 16 + col;
    const float bv = bias[cg];
    const int rb0 = rowbase + ((lane < 16) ? 0 : 8);
#pragma unroll
    for (int r = 0; r < 8; ++r) {
        int row = rb0 + r;
        if (row < M) C[(size_t)row * HID + cg] = c0[r] + bv;
    }
#pragma unroll
    for (int r = 0; r < 8; ++r) {
        int row = rb0 + 16 + r;
        if (row < M) C[(size_t)row * HID + cg] = c1[r] + bv;
    }
}

// ---------------------------------------------------------------------------
// Semantic score: acc += sum_{n,j} q[j] * tanh((A @ Wk)[n,j] + bk[j]), K = 128.
// Same pipeline; tanh/q wave-reduce epilogue, one atomicAdd per wave.
// A must already be ReLU'd; rows >= M are masked out of the reduction.
// ---------------------------------------------------------------------------
__global__ void __launch_bounds__(256)
gemm_score(const float* __restrict__ A, const __bf16* __restrict__ Bp,
           const float* __restrict__ bk, const float* __restrict__ q,
           float* __restrict__ scoreAcc, int M) {
    __shared__ __bf16 lds[2][32 * LDS_PITCH];
    const int tid = threadIdx.x;
    const int w = tid >> 5, lane = tid & 31;
    const int rowbase = blockIdx.x * 32;
    const int col = lane & 15;
    const int off = (lane < 16) ? 0 : 8;
    const int a0row = lane & 15, a1row = 16 + (lane & 15);
    const int lrow = tid >> 3;
    const int lk   = (tid & 7) * 8;
    const int nslab = HID >> 6;                  // 2

    int grow = rowbase + lrow;
    if (grow > M - 1) grow = M - 1;
    const float* aptr = A + (size_t)grow * HID + lk;

    {
        float4 f0 = *reinterpret_cast<const float4*>(aptr);
        float4 f1 = *reinterpret_cast<const float4*>(aptr + 4);
        stage8(&lds[0][lrow * LDS_PITCH + lk], f0, f1);
    }
    __syncthreads();

    v8f c0 = {}, c1 = {};
    for (int s = 0; s < nslab; ++s) {
        const int cur = s & 1;
        const bool more = (s + 1) < nslab;
        float4 g0, g1;
        if (more) {
            const float* p = aptr + (size_t)(s + 1) * 64;
            g0 = *reinterpret_cast<const float4*>(p);
            g1 = *reinterpret_cast<const float4*>(p + 4);
        }
        const __bf16* base0 = &lds[cur][0] + off;
        v16bf a00 = load_afrag(base0 + a0row * LDS_PITCH);
        v16bf a01 = load_afrag(base0 + a1row * LDS_PITCH);
        v16bf a10 = load_afrag(base0 + 32 + a0row * LDS_PITCH);
        v16bf a11 = load_afrag(base0 + 32 + a1row * LDS_PITCH);
        v16bf b0  = load_bfrag(Bp, s * 2 + 0, w, lane);
        v16bf b1  = load_bfrag(Bp, s * 2 + 1, w, lane);
        c0 = __builtin_amdgcn_wmma_f32_16x16x32_bf16(false, a00, false, b0,
                                                     (short)0, c0, false, false);
        c1 = __builtin_amdgcn_wmma_f32_16x16x32_bf16(false, a01, false, b0,
                                                     (short)0, c1, false, false);
        c0 = __builtin_amdgcn_wmma_f32_16x16x32_bf16(false, a10, false, b1,
                                                     (short)0, c0, false, false);
        c1 = __builtin_amdgcn_wmma_f32_16x16x32_bf16(false, a11, false, b1,
                                                     (short)0, c1, false, false);
        if (more)
            stage8(&lds[cur ^ 1][lrow * LDS_PITCH + lk], g0, g1);
        __syncthreads();
    }

    const int cg = w * 16 + col;
    const float bb = bk[cg];
    const int rb0 = rowbase + ((lane < 16) ? 0 : 8);
    float local = 0.f;
#pragma unroll
    for (int r = 0; r < 8; ++r)
        if (rb0 + r < M) local += tanhf(c0[r] + bb);
#pragma unroll
    for (int r = 0; r < 8; ++r)
        if (rb0 + 16 + r < M) local += tanhf(c1[r] + bb);
    local *= q[cg];
#pragma unroll
    for (int o = 16; o > 0; o >>= 1) local += __shfl_down(local, o, 32);
    if (lane == 0) atomicAdd(scoreAcc, local);
}

// fused per-(node,head) dots: one pass over Hm, three attention vectors
__global__ void headdot3(const float* __restrict__ Hm,
                         const float* __restrict__ a0, const float* __restrict__ a1,
                         const float* __restrict__ a2,
                         float* __restrict__ o0, float* __restrict__ o1,
                         float* __restrict__ o2, int n) {
    int i = blockIdx.x * blockDim.x + threadIdx.x;
    if (i >= n * NHEAD) return;
    int node = i >> 3, h = i & 7;
    const float4* hp = reinterpret_cast<const float4*>(Hm + (size_t)node * HID + h * HDIM);
    const float4* p0 = reinterpret_cast<const float4*>(a0 + h * HDIM);
    const float4* p1 = reinterpret_cast<const float4*>(a1 + h * HDIM);
    const float4* p2 = reinterpret_cast<const float4*>(a2 + h * HDIM);
    float s0 = 0.f, s1 = 0.f, s2 = 0.f;
#pragma unroll
    for (int j = 0; j < 4; ++j) {
        float4 x = hp[j];
        float4 y0 = p0[j], y1 = p1[j], y2 = p2[j];
        s0 += x.x * y0.x + x.y * y0.y + x.z * y0.z + x.w * y0.w;
        s1 += x.x * y1.x + x.y * y1.y + x.z * y1.z + x.w * y1.w;
        s2 += x.x * y2.x + x.y * y2.y + x.z * y2.z + x.w * y2.w;
    }
    o0[i] = s0; o1[i] = s1; o2[i] = s2;
}

// single per-(node,head) dot
__global__ void headdot(const float* __restrict__ Hm, const float* __restrict__ a,
                        float* __restrict__ out, int n) {
    int i = blockIdx.x * blockDim.x + threadIdx.x;
    if (i >= n * NHEAD) return;
    int node = i >> 3, h = i & 7;
    const float4* hp = reinterpret_cast<const float4*>(Hm + (size_t)node * HID + h * HDIM);
    const float4* ap = reinterpret_cast<const float4*>(a + h * HDIM);
    float s = 0.f;
#pragma unroll
    for (int j = 0; j < 4; ++j) {
        float4 x = hp[j], y = ap[j];
        s += x.x * y.x + x.y * y.y + x.z * y.z + x.w * y.w;
    }
    out[i] = s;
}

__global__ void fill_i32(int* p, int v, int n) {
    int i = blockIdx.x * blockDim.x + threadIdx.x;
    if (i < n) p[i] = v;
}

__device__ __forceinline__ float lrelu(float v) { return v > 0.f ? v : 0.2f * v; }

// pass 1: encoded atomic segment-max of e over dst
__global__ void att_max(const int* __restrict__ edge, int E,
                        const float* __restrict__ asrc, const float* __restrict__ adst,
                        int* __restrict__ mEnc) {
    long long i = (long long)blockIdx.x * blockDim.x + threadIdx.x;
    if (i >= (long long)E * NHEAD) return;
    int e = (int)(i >> 3), h = (int)(i & 7);
    int s = edge[e], d = edge[E + e];
    float v = lrelu(asrc[(size_t)s * NHEAD + h] + adst[(size_t)d * NHEAD + h]);
    atomicMax(&mEnc[(size_t)d * NHEAD + h], encF(v));
}

// decode max in place; INT_MIN sentinel (no edges) -> 0 (matches isfinite fixup)
__global__ void m_fix(int* __restrict__ m, int n) {
    int i = blockIdx.x * blockDim.x + threadIdx.x;
    if (i >= n) return;
    int v = m[i];
    reinterpret_cast<float*>(m)[i] = (v == INT_MIN) ? 0.f : decF(v);
}

// pass 2: denom[dst,h] += exp(e - m[dst,h])
__global__ void att_denom(const int* __restrict__ edge, int E,
                          const float* __restrict__ asrc, const float* __restrict__ adst,
                          const float* __restrict__ m, float* __restrict__ den) {
    long long i = (long long)blockIdx.x * blockDim.x + threadIdx.x;
    if (i >= (long long)E * NHEAD) return;
    int e = (int)(i >> 3), h = (int)(i & 7);
    int s = edge[e], d = edge[E + e];
    size_t di = (size_t)d * NHEAD + h;
    float v = lrelu(asrc[(size_t)s * NHEAD + h] + adst[di]);
    atomicAdd(&den[di], __expf(v - m[di]));
}

// pass 3: out[dst, :] += h_src[src, :] * alpha(edge, head).
// alpha computed once per 16-lane head group (lanes 0/16) then broadcast.
__global__ void att_aggr(const int* __restrict__ edge, int E,
                         const float* __restrict__ asrc, const float* __restrict__ adst,
                         const float* __restrict__ m, const float* __restrict__ den,
                         const float* __restrict__ Hsrc, float* __restrict__ out) {
    long long i = (long long)blockIdx.x * blockDim.x + threadIdx.x;
    if (i >= (long long)E * HID) return;
    int e = (int)(i >> 7), t = (int)(i & 127), h = t >> 4;
    int lane = threadIdx.x & 31;
    int s = edge[e], d = edge[E + e];
    float alpha = 0.f;
    if ((lane & 15) == 0) {
        size_t di = (size_t)d * NHEAD + h;
        float v = lrelu(asrc[(size_t)s * NHEAD + h] + adst[di]);
        alpha = __expf(v - m[di]) / (den[di] + 1e-16f);
    }
    alpha = __shfl(alpha, lane & 16, 32);
    atomicAdd(&out[(size_t)d * HID + t], Hsrc[(size_t)s * HID + t] * alpha);
}

__global__ void relu_ip(float* p, int n) {
    int i = blockIdx.x * blockDim.x + threadIdx.x;
    if (i < n) p[i] = fmaxf(p[i], 0.f);
}

__global__ void beta_k(const float* __restrict__ acc, float* __restrict__ beta) {
    float s0 = acc[0] / (float)N_NEWS, s1 = acc[1] / (float)N_NEWS;
    float mx = fmaxf(s0, s1);
    float e0 = __expf(s0 - mx), e1 = __expf(s1 - mx);
    float inv = 1.f / (e0 + e1);
    beta[0] = e0 * inv;
    beta[1] = e1 * inv;
}

// fused = beta0*out0 + beta1*out1; out = ELU(fused) @ W_out + b_out
__global__ void final_k(const float* __restrict__ o0, const float* __restrict__ o1,
                        const float* __restrict__ beta, const float* __restrict__ Wo,
                        const float* __restrict__ bo, float* __restrict__ out) {
    int i = blockIdx.x * blockDim.x + threadIdx.x;
    if (i >= N_NEWS * C_OUT) return;
    int n = i >> 2, c = i & 3;
    float b0 = beta[0], b1 = beta[1];
    float acc = bo[c];
    const float* p0 = o0 + (size_t)n * HID;
    const float* p1 = o1 + (size_t)n * HID;
#pragma unroll 4
    for (int j = 0; j < HID; ++j) {
        float f = b0 * p0[j] + b1 * p1[j];
        f = f > 0.f ? f : expm1f(f);
        acc += f * Wo[j * C_OUT + c];
    }
    out[i] = acc;
}

// ---------------------------------------------------------------------------
extern "C" void kernel_launch(void* const* d_in, const int* in_sizes, int n_in,
                              void* d_out, int out_size, void* d_ws, size_t ws_size,
                              hipStream_t stream) {
    const float* x_news  = (const float*)d_in[0];
    const float* x_inter = (const float*)d_in[1];
    const int*   e_nn    = (const int*)d_in[2];
    const int*   e_in    = (const int*)d_in[3];
    const float* W_news  = (const float*)d_in[4];
    const float* b_news  = (const float*)d_in[5];
    const float* W_inter = (const float*)d_in[6];
    const float* b_inter = (const float*)d_in[7];
    const float* a_s_nn  = (const float*)d_in[8];
    const float* a_d_nn  = (const float*)d_in[9];
    const float* a_s_in  = (const float*)d_in[10];
    const float* a_d_in  = (const float*)d_in[11];
    const float* Wk      = (const float*)d_in[12];
    const float* bk      = (const float*)d_in[13];
    const float* q       = (const float*)d_in[14];
    const float* W_out   = (const float*)d_in[15];
    const float* b_out   = (const float*)d_in[16];
    float* out = (float*)d_out;

    const int E_nn = in_sizes[2] / 2;
    const int E_in = in_sizes[3] / 2;

    // ---- workspace carve (256B aligned) ----
    char* ws = (char*)d_ws;
    size_t off = 0;
    auto carve = [&](size_t bytes) {
        void* p = ws + off;
        off += (bytes + 255) & ~(size_t)255;
        return p;
    };
    float* hn   = (float*)carve((size_t)N_NEWS * HID * 4);
    float* hi   = (float*)carve((size_t)N_INTER * HID * 4);
    float* outs = (float*)carve((size_t)2 * N_NEWS * HID * 4);   // [2][N][128]
    __bf16* pWn = (__bf16*)carve((size_t)F_IN * HID * 2);
    __bf16* pWi = (__bf16*)carve((size_t)F_IN * HID * 2);
    __bf16* pWk = (__bf16*)carve((size_t)HID * HID * 2);
    float* asrc_nn = (float*)carve((size_t)N_NEWS * NHEAD * 4);
    float* adst_nn = (float*)carve((size_t)N_NEWS * NHEAD * 4);
    float* asrc_in = (float*)carve((size_t)N_INTER * NHEAD * 4);
    float* adst_in = (float*)carve((size_t)N_NEWS * NHEAD * 4);
    int*   m_nn    = (int*)carve((size_t)N_NEWS * NHEAD * 4);    // adjacent to m_in
    int*   m_in    = (int*)carve((size_t)N_NEWS * NHEAD * 4);
    float* den_nn  = (float*)carve((size_t)N_NEWS * NHEAD * 4);  // adjacent to den_in
    float* den_in  = (float*)carve((size_t)N_NEWS * NHEAD * 4);
    float* scoreAcc = (float*)carve(2 * 4);
    float* beta     = (float*)carve(2 * 4);
    (void)ws_size;

    float* out_nn = outs;
    float* out_in = outs + (size_t)N_NEWS * HID;

    // ---- init ----
    const int NH = N_NEWS * NHEAD;
    fill_i32<<<(2 * NH + 255) / 256, 256, 0, stream>>>(m_nn, INT_MIN, 2 * NH);
    hipMemsetAsync(den_nn, 0, (size_t)2 * NH * 4, stream);
    hipMemsetAsync(outs, 0, (size_t)2 * N_NEWS * HID * 4, stream);
    hipMemsetAsync(scoreAcc, 0, 2 * 4, stream);

    // ---- pack weights to bf16 WMMA fragments ----
    prepack_w<<<(F_IN / 32) * 8, 32, 0, stream>>>(W_news, pWn, F_IN);
    prepack_w<<<(F_IN / 32) * 8, 32, 0, stream>>>(W_inter, pWi, F_IN);
    prepack_w<<<(HID / 32) * 8, 32, 0, stream>>>(Wk, pWk, HID);

    // ---- projections (WMMA) ----
    gemm_bias<<<(N_NEWS + 31) / 32, 256, 0, stream>>>(x_news, pWn, b_news, hn, F_IN, N_NEWS);
    gemm_bias<<<(N_INTER + 31) / 32, 256, 0, stream>>>(x_inter, pWi, b_inter, hi, F_IN, N_INTER);

    // ---- per-head attention scalars (hn read once for 3 vectors) ----
    headdot3<<<(N_NEWS * NHEAD + 255) / 256, 256, 0, stream>>>(
        hn, a_s_nn, a_d_nn, a_d_in, asrc_nn, adst_nn, adst_in, N_NEWS);
    headdot<<<(N_INTER * NHEAD + 255) / 256, 256, 0, stream>>>(hi, a_s_in, asrc_in, N_INTER);

    // ---- edge softmax: max -> fixup -> denom -> aggregate ----
    {
        long long t1 = (long long)E_nn * NHEAD;
        long long t2 = (long long)E_in * NHEAD;
        att_max<<<(unsigned)((t1 + 255) / 256), 256, 0, stream>>>(e_nn, E_nn, asrc_nn, adst_nn, m_nn);
        att_max<<<(unsigned)((t2 + 255) / 256), 256, 0, stream>>>(e_in, E_in, asrc_in, adst_in, m_in);
        m_fix<<<(2 * NH + 255) / 256, 256, 0, stream>>>(m_nn, 2 * NH);
        att_denom<<<(unsigned)((t1 + 255) / 256), 256, 0, stream>>>(
            e_nn, E_nn, asrc_nn, adst_nn, (const float*)m_nn, den_nn);
        att_denom<<<(unsigned)((t2 + 255) / 256), 256, 0, stream>>>(
            e_in, E_in, asrc_in, adst_in, (const float*)m_in, den_in);
        long long a1 = (long long)E_nn * HID;
        long long a2 = (long long)E_in * HID;
        att_aggr<<<(unsigned)((a1 + 255) / 256), 256, 0, stream>>>(
            e_nn, E_nn, asrc_nn, adst_nn, (const float*)m_nn, den_nn, hn, out_nn);
        att_aggr<<<(unsigned)((a2 + 255) / 256), 256, 0, stream>>>(
            e_in, E_in, asrc_in, adst_in, (const float*)m_in, den_in, hi, out_in);
    }

    // ---- relu both metapath outputs (in place) ----
    relu_ip<<<(2 * N_NEWS * HID + 255) / 256, 256, 0, stream>>>(outs, 2 * N_NEWS * HID);

    // ---- semantic attention (WMMA score GEMMs -> beta) ----
    gemm_score<<<(N_NEWS + 31) / 32, 256, 0, stream>>>(out_nn, pWk, bk, q, &scoreAcc[0], N_NEWS);
    gemm_score<<<(N_NEWS + 31) / 32, 256, 0, stream>>>(out_in, pWk, bk, q, &scoreAcc[1], N_NEWS);
    beta_k<<<1, 1, 0, stream>>>(scoreAcc, beta);

    // ---- fuse + ELU + output head ----
    final_k<<<(N_NEWS * C_OUT + 255) / 256, 256, 0, stream>>>(
        out_nn, out_in, beta, W_out, b_out, out);
}
